// MatrixLSTM_2224793060032
// MI455X (gfx1250) — compile-verified, matrix operand order checked by reading, not run on hardware
//
#include <hip/hip_runtime.h>
#include <math.h>

// ---------------------------------------------------------------------------
// Types for CDNA5 WMMA
// ---------------------------------------------------------------------------
typedef __attribute__((ext_vector_type(16))) __bf16 v16bf;
typedef __attribute__((ext_vector_type(8)))  float  v8f;

// Problem constants (match reference)
#define BB 4
#define DD 512
#define HH 512
#define OO 512
#define NGATE 6
#define NTOT  (NGATE * HH)       // 3072 concatenated gate columns
#define NTOT7 ((NGATE + 1) * HH) // + W_out

// ---------------------------------------------------------------------------
// CDNA5 async-copy helpers (ASYNCcnt-tracked global->LDS DMA)
// ---------------------------------------------------------------------------
__device__ inline void async_copy_b128(unsigned lds_dst, const float* src) {
    // GLOBAL_LOAD_ASYNC_TO_LDS_B128: VDST = LDS byte address, VADDR = 64-bit global
    asm volatile("global_load_async_to_lds_b128 %0, %1, off"
                 :: "v"(lds_dst), "v"(src)
                 : "memory");
}

template <int N>
__device__ inline void wait_asynccnt() {
    asm volatile("s_wait_asynccnt %0" :: "n"(N) : "memory");
}

// Generic pointer to a __shared__ object: low 32 bits are the LDS byte address
__device__ inline unsigned lds_addr(const void* p) {
    return (unsigned)(uintptr_t)p;
}

// ---------------------------------------------------------------------------
// Fragment loader: 16 bf16 elements per lane, CDNA5 16-bit A/B striping.
// Elements 0..7  -> K = kbase + kg*8 + 0..7      (contiguous 16 bytes)
// Elements 8..15 -> K = kbase + 16 + kg*8 + 0..7 (contiguous 16 bytes)
// ---------------------------------------------------------------------------
__device__ inline v16bf load_bf16_frag(const __bf16* p) {
    union { v16bf v; uint4 u[2]; } r;
    r.u[0] = *(const uint4*)(p);
    r.u[1] = *(const uint4*)(p + 16);
    return r.v;
}

// ---------------------------------------------------------------------------
// Conversion / packing kernels
// ---------------------------------------------------------------------------
__global__ void k_convert_x(const float* __restrict__ x, __bf16* __restrict__ xbf, int n) {
    int i = blockIdx.x * blockDim.x + threadIdx.x;
    if (i < n) xbf[i] = (__bf16)x[i];
}

__global__ void k_pack_w(const float* __restrict__ Wq, const float* __restrict__ Wk,
                         const float* __restrict__ Wv, const float* __restrict__ Wf,
                         const float* __restrict__ Wi, const float* __restrict__ Wo,
                         const float* __restrict__ Wout, __bf16* __restrict__ Wt) {
    int idx = blockIdx.x * blockDim.x + threadIdx.x;   // n_global*512 + k
    int ng = idx >> 9;          // 0..3583
    int k  = idx & 511;
    int g  = ng >> 9;           // which matrix
    int n  = ng & 511;
    const float* W;
    switch (g) {
        case 0: W = Wq; break; case 1: W = Wk; break; case 2: W = Wv; break;
        case 3: W = Wf; break; case 4: W = Wi; break; case 5: W = Wo; break;
        default: W = Wout; break;
    }
    Wt[(size_t)ng * DD + k] = (__bf16)W[(size_t)k * HH + n];
}

__global__ void k_pack_b(const float* __restrict__ bq, const float* __restrict__ bk,
                         const float* __restrict__ bv, const float* __restrict__ bf,
                         const float* __restrict__ bi, const float* __restrict__ bo,
                         const float* __restrict__ bout, float* __restrict__ bias) {
    int ng = blockIdx.x * blockDim.x + threadIdx.x;
    if (ng >= NTOT7) return;
    int g = ng >> 9, n = ng & 511;
    const float* b;
    switch (g) {
        case 0: b = bq; break; case 1: b = bk; break; case 2: b = bv; break;
        case 3: b = bf; break; case 4: b = bi; break; case 5: b = bo; break;
        default: b = bout; break;
    }
    bias[ng] = b[n];
}

// ---------------------------------------------------------------------------
// WMMA GEMM: out[M, ldo] = A[M,K](bf16) * Wt[N,K](bf16, N-major) + bias
// Block = 256 threads = 8 waves. Block tile: 128(M) x 64(N).
// Wave tile: 16(M) x 64(N) = 4 accumulators; K loop in chunks of 32.
// mode 0: gate epilogue (gate1: *1/sqrt(512); gate5: sigmoid)
// mode 1: plain bias epilogue
// ---------------------------------------------------------------------------
__global__ void k_gemm_wmma(const __bf16* __restrict__ A, const __bf16* __restrict__ Wt,
                            const float* __restrict__ bias, float* __restrict__ out,
                            int K, int ldo, int mode) {
    const int n0   = blockIdx.x * 64;
    const int m0   = blockIdx.y * 128;
    const int wave = threadIdx.x >> 5;
    const int lane = threadIdx.x & 31;
    const int m_l  = lane & 15;
    const int kg   = lane >> 4;
    const int mrow = m0 + wave * 16;

    v8f acc0 = {}, acc1 = {}, acc2 = {}, acc3 = {};

    const __bf16* Arow = A + (size_t)(mrow + m_l) * K;
    const __bf16* B0 = Wt + (size_t)(n0 +  0 + m_l) * K;
    const __bf16* B1 = Wt + (size_t)(n0 + 16 + m_l) * K;
    const __bf16* B2 = Wt + (size_t)(n0 + 32 + m_l) * K;
    const __bf16* B3 = Wt + (size_t)(n0 + 48 + m_l) * K;

    for (int kb = 0; kb < K; kb += 32) {
        const int ko = kb + kg * 8;
        v16bf a  = load_bf16_frag(Arow + ko);
        v16bf b0 = load_bf16_frag(B0 + ko);
        v16bf b1 = load_bf16_frag(B1 + ko);
        v16bf b2 = load_bf16_frag(B2 + ko);
        v16bf b3 = load_bf16_frag(B3 + ko);
        acc0 = __builtin_amdgcn_wmma_f32_16x16x32_bf16(false, a, false, b0, (short)0, acc0, false, false);
        acc1 = __builtin_amdgcn_wmma_f32_16x16x32_bf16(false, a, false, b1, (short)0, acc1, false, false);
        acc2 = __builtin_amdgcn_wmma_f32_16x16x32_bf16(false, a, false, b2, (short)0, acc2, false, false);
        acc3 = __builtin_amdgcn_wmma_f32_16x16x32_bf16(false, a, false, b3, (short)0, acc3, false, false);
    }

    v8f accs[4] = {acc0, acc1, acc2, acc3};
    #pragma unroll
    for (int j = 0; j < 4; ++j) {
        const int n  = n0 + j * 16 + m_l;
        const float bv = bias[n];
        const int gate = n >> 9;
        #pragma unroll
        for (int v = 0; v < 8; ++v) {
            const int m = mrow + v + 8 * kg;
            float val = accs[j][v] + bv;
            if (mode == 0) {
                if (gate == 1)      val *= 0.04419417382415922f;      // 1/sqrt(512)
                else if (gate == 5) val = 1.0f / (1.0f + expf(-val)); // sigmoid(o)
            }
            out[(size_t)m * ldo + n] = val;
        }
    }
}

// ---------------------------------------------------------------------------
// Sequential mLSTM scan with double-buffered async global->LDS gate staging.
// Grid: 32 blocks = 4 batches x 8 row-blocks (64 rows of C each).
// 256 threads = 8 waves; each wave keeps 8 rows x 512 cols of C in registers.
// Per step: issue async copy of the NEXT 12KB gate row, wait asynccnt<=3 so
// the CURRENT row is resident, then do the VALU sweep out of LDS.
// ---------------------------------------------------------------------------
__global__ void k_scan(const float* __restrict__ gates,  // [B*T, 3072] q|k|v|f|i|o
                       __bf16* __restrict__ h_bf,        // [B*T, 512]
                       float* __restrict__ out_C,        // [B, 512, 512]
                       float* __restrict__ out_n,        // [B, 512]
                       float* __restrict__ out_m,        // [B, 512]
                       int T) {
    const int b    = blockIdx.x >> 3;
    const int rb   = blockIdx.x & 7;
    const int row0 = rb * 64;
    const int tid  = threadIdx.x;
    const int wave = tid >> 5;
    const int lane = tid & 31;

    __shared__ float s_gbuf[2][NTOT];     // 2 x 12KB double buffer
    __shared__ float s_f[HH], s_i[HH];
    __shared__ float s_red[8];
    __shared__ float s_nq;

    // Issue: 3 x b128 per thread covers one 3072-float gate row
    auto issue_row = [&](int buf, int t) {
        const float* g = gates + (size_t)(b * T + t) * NTOT;
        #pragma unroll
        for (int s = 0; s < 3; ++s) {
            const int e = s * 1024 + tid * 4;
            async_copy_b128(lds_addr(&s_gbuf[buf][e]), g + e);
        }
    };

    float m_s[2] = {0.f, 0.f};
    float n_s[2] = {0.f, 0.f};
    float Creg[8][16];
    #pragma unroll
    for (int r = 0; r < 8; ++r)
        #pragma unroll
        for (int c = 0; c < 16; ++c) Creg[r][c] = 0.f;

    issue_row(0, 0);   // prologue

    for (int t = 0; t < T; ++t) {
        const int cur = t & 1;
        if (t + 1 < T) {
            issue_row(cur ^ 1, t + 1);
            wait_asynccnt<3>();           // 3 newest in flight = next row; current done
        } else {
            wait_asynccnt<0>();
        }
        __syncthreads();                  // [B1] all waves' async copies visible
        const float* gb = s_gbuf[cur];

        // Elementwise gate recurrence for this thread's 2 hidden indices
        float nq_part = 0.f;
        #pragma unroll
        for (int j = 0; j < 2; ++j) {
            const int idx = tid + j * 256;
            const float qv = gb[0 * HH + idx];
            const float kv = gb[1 * HH + idx];
            const float fl = gb[3 * HH + idx];
            const float il = gb[4 * HH + idx];
            const float mn = fmaxf(fl + m_s[j], il);
            const float fe = expf(fl + m_s[j] - mn);
            const float ie = expf(il - mn);
            m_s[j] = mn;
            n_s[j] = fe * n_s[j] + ie * kv;
            nq_part += n_s[j] * qv;
            s_f[idx] = fe; s_i[idx] = ie;
        }
        // Wave-level reduction of n.q, then stage per-wave partials
        float r = nq_part;
        #pragma unroll
        for (int off = 16; off > 0; off >>= 1) r += __shfl_xor(r, off, 32);
        if (lane == 0) s_red[wave] = r;
        __syncthreads();                  // [B2] s_f/s_i/s_red visible

        if (tid == 0) {
            float s = 0.f;
            #pragma unroll
            for (int w = 0; w < 8; ++w) s += s_red[w];
            s_nq = fabsf(s) + 1e-8f;
        }
        __syncthreads();                  // [B3] s_nq visible
        const float n_q = s_nq;

        // Per-lane column slices of k, q from the staged gate row
        float kk[16], qq[16];
        #pragma unroll
        for (int cc = 0; cc < 16; ++cc) {
            const int c = lane + 32 * cc;
            qq[cc] = gb[0 * HH + c];
            kk[cc] = gb[1 * HH + c];
        }

        // C update + C.q for this wave's 8 rows
        float cq[8];
        #pragma unroll
        for (int rr = 0; rr < 8; ++rr) {
            const int row = row0 + wave * 8 + rr;
            const float fr  = s_f[row];
            const float air = s_i[row] * gb[2 * HH + row];   // i * v
            float acc = 0.f;
            #pragma unroll
            for (int cc = 0; cc < 16; ++cc) {
                Creg[rr][cc] = fr * Creg[rr][cc] + air * kk[cc];
                acc += Creg[rr][cc] * qq[cc];
            }
            #pragma unroll
            for (int off = 16; off > 0; off >>= 1) acc += __shfl_xor(acc, off, 32);
            cq[rr] = acc;
        }

        // h = o * (C.q) / |n.q|  -> bf16 activations for the output GEMM
        if (lane < 8) {
            const int row = row0 + wave * 8 + lane;
            const float ov = gb[5 * HH + row];               // sigmoid applied in GEMM epilogue
            const float hv = ov * cq[lane] / n_q;
            h_bf[(size_t)(b * T + t) * HH + row] = (__bf16)hv;
        }
        // Next iteration's [B1] guards s_f/s_i reuse; its refill of this gb buffer is issued before [B1].
    }

    // Final state writeback
    #pragma unroll
    for (int rr = 0; rr < 8; ++rr) {
        const int row = row0 + wave * 8 + rr;
        #pragma unroll
        for (int cc = 0; cc < 16; ++cc)
            out_C[((size_t)b * HH + row) * HH + (lane + 32 * cc)] = Creg[rr][cc];
    }
    #pragma unroll
    for (int j = 0; j < 2; ++j) {
        const int idx = tid + j * 256;
        out_n[b * HH + idx] = n_s[j];
        out_m[b * HH + idx] = m_s[j];
    }
}

// ---------------------------------------------------------------------------
// Launcher
// ---------------------------------------------------------------------------
extern "C" void kernel_launch(void* const* d_in, const int* in_sizes, int n_in,
                              void* d_out, int out_size, void* d_ws, size_t ws_size,
                              hipStream_t stream) {
    const float* x    = (const float*)d_in[0];
    const float* Wq   = (const float*)d_in[1];  const float* bq = (const float*)d_in[2];
    const float* Wk   = (const float*)d_in[3];  const float* bk = (const float*)d_in[4];
    const float* Wv   = (const float*)d_in[5];  const float* bv = (const float*)d_in[6];
    const float* Wf   = (const float*)d_in[7];  const float* bf = (const float*)d_in[8];
    const float* Wi   = (const float*)d_in[9];  const float* bi = (const float*)d_in[10];
    const float* Wo   = (const float*)d_in[11]; const float* bo = (const float*)d_in[12];
    const float* Wout = (const float*)d_in[13]; const float* bout = (const float*)d_in[14];

    const int T = in_sizes[0] / (BB * DD);   // 2048
    const int M = BB * T;                    // 8192

    // Workspace carving (256B aligned)
    char* ws = (char*)d_ws;
    size_t off = 0;
    auto take = [&](size_t bytes) -> char* {
        char* p = ws + off;
        off = (off + bytes + 255) & ~(size_t)255;
        return p;
    };
    __bf16* xbf   = (__bf16*)take((size_t)M * DD * sizeof(__bf16));
    __bf16* Wtbf  = (__bf16*)take((size_t)NTOT7 * DD * sizeof(__bf16));
    float*  bias  = (float*)take((size_t)NTOT7 * sizeof(float));
    float*  gbuf  = (float*)take((size_t)M * NTOT * sizeof(float));
    __bf16* hbf   = (__bf16*)take((size_t)M * HH * sizeof(__bf16));

    float* out   = (float*)d_out;
    float* out_f = out;                               // [B,T,O]
    float* out_C = out_f + (size_t)M * OO;            // [B,H,H]
    float* out_n = out_C + (size_t)BB * HH * HH;      // [B,H]
    float* out_m = out_n + (size_t)BB * HH;           // [B,H]

    // 1) Convert / pack
    k_convert_x<<<(M * DD + 255) / 256, 256, 0, stream>>>(x, xbf, M * DD);
    k_pack_w<<<(NTOT7 * DD + 255) / 256, 256, 0, stream>>>(Wq, Wk, Wv, Wf, Wi, Wo, Wout, Wtbf);
    k_pack_b<<<(NTOT7 + 255) / 256, 256, 0, stream>>>(bq, bk, bv, bf, bi, bo, bout, bias);

    // 2) Fused 6-gate projection GEMM (WMMA bf16)
    {
        dim3 grid(NTOT / 64, M / 128);
        k_gemm_wmma<<<grid, 256, 0, stream>>>(xbf, Wtbf, bias, gbuf, DD, NTOT, 0);
    }

    // 3) Sequential mLSTM scan (C resident in VGPRs, async LDS gate pipeline)
    k_scan<<<BB * 8, 256, 0, stream>>>(gbuf, hbf, out_C, out_n, out_m, T);

    // 4) Output projection GEMM (WMMA bf16)
    {
        dim3 grid(OO / 64, M / 128);
        k_gemm_wmma<<<grid, 256, 0, stream>>>(hbf, Wtbf + (size_t)NGATE * HH * DD,
                                              bias + NGATE * HH, out_f, HH, OO, 1);
    }
}
